// ISSB_28578712388132
// MI455X (gfx1250) — matrix operand order repaired and checked by reading, not compiled, not run on hardware
//
#include <hip/hip_runtime.h>
#include <hip/hip_bf16.h>

// ---------------- CDNA5 WMMA helpers (wave32, 16x16x32 f16 -> f32) ----------
typedef __attribute__((ext_vector_type(16))) _Float16 v16h;
typedef __attribute__((ext_vector_type(8)))  _Float16 v8h;
typedef __attribute__((ext_vector_type(8)))  float    v8f;

#define EPSV 1e-5f

// A fragment: 16(M) x 32(K) f16, row-major in LDS with element stride lda.
// Per-lane data is one contiguous 16-halfword run (matches B):
// K = half*16..half*16+15  -> two ds_load_b128.
__device__ inline v16h load_a_frag(const _Float16* A, int lda) {
  const int lane = threadIdx.x & 31;
  const int m = lane & 15, half = lane >> 4;
  const _Float16* row = A + m * lda + half * 16;
  v8h lo = *(const v8h*)(row);        // K = half*16 + 0..7
  v8h hi = *(const v8h*)(row + 8);    // K = half*16 + 8..15
  return __builtin_shufflevector(lo, hi, 0, 1, 2, 3, 4, 5, 6, 7,
                                 8, 9, 10, 11, 12, 13, 14, 15);
}

// B fragment from an N-major tile Bt[N][K] (K contiguous, stride ldk).
// Lane (n, half) needs Bt[n][half*16 .. half*16+15]: one aligned 32B read.
__device__ inline v16h load_b_frag_t(const _Float16* Bt, int ldk) {
  const int lane = threadIdx.x & 31;
  const int n = lane & 15, half = lane >> 4;
  return *(const v16h*)(Bt + n * ldk + half * 16);
}

__device__ inline v8f wmma_f16(v16h a, v16h b, v8f c) {
  return __builtin_amdgcn_wmma_f32_16x16x32_f16(false, a, false, b, (short)0, c,
                                                false, false);
}

__device__ inline float silu_f(float x) { return x / (1.0f + __expf(-x)); }
__device__ inline float softplus_f(float x) {
  return (x > 20.0f) ? x : __logf(1.0f + __expf(x));
}

// scan index p -> spatial index (h*48+w) for direction k (H=W=48, L=2304)
__device__ inline int scan_pos(int k, int p) {
  switch (k) {
    case 0:  return p;
    case 1:  return (p % 48) * 48 + (p / 48);
    case 2:  return 2303 - p;
    default: { int q = 2303 - p; return (q % 48) * 48 + (q / 48); }
  }
}

// ---------------- Kernel 1: LayerNorm(C=96) + in_proj GEMM ------------------
// xz[m][e] = sum_c LN(x)[m][c] * Wip[e][c],  m in [0,4608), e in [0,384)
__global__ void k_ln_inproj(const float* __restrict__ x,      // (B,96,48,48)
                            const float* __restrict__ ln_g,
                            const float* __restrict__ ln_b,
                            const float* __restrict__ wip,    // (384,96)
                            float* __restrict__ xz) {         // (4608,384)
  __shared__ float Araw[16][96];
  __shared__ alignas(32) _Float16 Ah[16][96];
  __shared__ alignas(32) _Float16 Bs[64][96];   // N-major: Bs[n][k]
  __shared__ float s_mean[16], s_rinv[16];

  const int tid = threadIdx.x;                // 128 threads
  const int m0  = blockIdx.x * 16;            // 288 blocks
  const int n0  = blockIdx.y * 64;            // 6 blocks

  __builtin_prefetch(&wip[n0 * 96], 0, 3);    // global_prefetch_b8

  for (int i = tid; i < 16 * 96; i += 128) {
    const int r = i / 96, c = i % 96;
    const int m = m0 + r, b = m / 2304, l = m % 2304;
    Araw[r][c] = x[((size_t)b * 96 + c) * 2304 + l];
  }
  __syncthreads();
  if (tid < 16) {
    float s = 0.f;
#pragma unroll 8
    for (int c = 0; c < 96; ++c) s += Araw[tid][c];
    const float mean = s * (1.0f / 96.0f);
    float v = 0.f;
#pragma unroll 8
    for (int c = 0; c < 96; ++c) { float d = Araw[tid][c] - mean; v += d * d; }
    s_mean[tid] = mean;
    s_rinv[tid] = rsqrtf(v * (1.0f / 96.0f) + EPSV);
  }
  __syncthreads();
  for (int i = tid; i < 16 * 96; i += 128) {
    const int r = i / 96, c = i % 96;
    Ah[r][c] = (_Float16)((Araw[r][c] - s_mean[r]) * s_rinv[r] * ln_g[c] + ln_b[c]);
  }
  for (int i = tid; i < 64 * 96; i += 128) {   // straight copy: W is [e][c]
    const int j = i / 96, kk = i % 96;
    Bs[j][kk] = (_Float16)wip[(size_t)(n0 + j) * 96 + kk];
  }
  __syncthreads();

  const int w = tid >> 5;                     // 4 waves, 16 N-cols each
  v8f acc = {};
#pragma unroll
  for (int kt = 0; kt < 3; ++kt) {
    v16h a = load_a_frag(&Ah[0][kt * 32], 96);
    v16h b = load_b_frag_t(&Bs[w * 16][kt * 32], 96);
    acc = wmma_f16(a, b, acc);
  }
  const int lane = tid & 31, n = lane & 15, half = lane >> 4;
#pragma unroll
  for (int r = 0; r < 8; ++r)
    xz[(size_t)(m0 + half * 8 + r) * 384 + n0 + w * 16 + n] = acc[r];
}

// ---------------- Kernel 2: depthwise 3x3 conv + bias + SiLU ----------------
__global__ void k_dwconv_silu(const float* __restrict__ xz,   // (4608,384), ch 0..191
                              const float* __restrict__ dw,   // (192,1,3,3)
                              const float* __restrict__ db,
                              float* __restrict__ xc) {       // (B,192,2304)
  const int idx = blockIdx.x * 256 + threadIdx.x;
  if (idx >= 2 * 192 * 2304) return;
  const int b = idx / (192 * 2304);
  const int d = (idx / 2304) % 192;
  const int l = idx % 2304;
  const int h = l / 48, wq = l % 48;
  float s = db[d];
#pragma unroll
  for (int ty = 0; ty < 3; ++ty) {
#pragma unroll
    for (int tx = 0; tx < 3; ++tx) {
      const int iy = h + ty - 1, ix = wq + tx - 1;
      if (iy >= 0 && iy < 48 && ix >= 0 && ix < 48)
        s += dw[d * 9 + ty * 3 + tx] *
             xz[((size_t)b * 2304 + iy * 48 + ix) * 384 + d];
    }
  }
  xc[((size_t)b * 192 + d) * 2304 + l] = silu_f(s);
}

// ---------------- Kernel 3: x_proj GEMM per direction -----------------------
// x_dbl[b,k,l,c] = sum_d xs[b,k,d,l] * Wxp[k,c,d], c in [0,38), padded ld 40
__global__ void k_xdbl(const float* __restrict__ xc,          // (B,192,2304)
                       const float* __restrict__ wxp,         // (4,38,192)
                       float* __restrict__ x_dbl) {           // (8,2304,40)
  __shared__ alignas(32) _Float16 Ah[16][192];
  __shared__ alignas(32) _Float16 Bs[48][192];  // N-major: Bs[c][d]
  const int tid = threadIdx.x;                // 96 threads, 3 waves
  const int l0  = blockIdx.x * 16;            // 144
  const int bk  = blockIdx.y;                 // 8
  const int b = bk >> 2, k = bk & 3;

  for (int i = tid; i < 16 * 192; i += 96) {
    const int r = i / 192, dd = i % 192;
    Ah[r][dd] = (_Float16)xc[((size_t)b * 192 + dd) * 2304 + scan_pos(k, l0 + r)];
  }
  for (int i = tid; i < 48 * 192; i += 96) {   // straight copy: Wxp is [k][c][d]
    const int c = i / 192, dd = i % 192;
    Bs[c][dd] = (c < 38) ? (_Float16)wxp[((size_t)k * 38 + c) * 192 + dd]
                         : (_Float16)0.0f;
  }
  __syncthreads();

  const int w = tid >> 5;
  v8f acc = {};
#pragma unroll
  for (int kt = 0; kt < 6; ++kt) {
    v16h a  = load_a_frag(&Ah[0][kt * 32], 192);
    v16h bf = load_b_frag_t(&Bs[w * 16][kt * 32], 192);
    acc = wmma_f16(a, bf, acc);
  }
  const int lane = tid & 31, n = lane & 15, half = lane >> 4;
  const int c = w * 16 + n;
  if (c < 38) {
#pragma unroll
    for (int r = 0; r < 8; ++r)
      x_dbl[((size_t)bk * 2304 + l0 + half * 8 + r) * 40 + c] = acc[r];
  }
}

// ---------------- Kernel 4: selective scan (sequential over L) --------------
// one block per (b,k); one thread per d; 16 states in registers
__global__ void k_scan(const float* __restrict__ xc,          // (B,192,2304)
                       const float* __restrict__ x_dbl,       // (8,2304,40)
                       const float* __restrict__ dt_w,        // (4,192,6)
                       const float* __restrict__ dt_b,        // (4,192)
                       const float* __restrict__ A_logs,      // (4,192,16)
                       const float* __restrict__ Ds,          // (4,192)
                       float* __restrict__ ys) {              // (8,192,2304) scan order
  __shared__ float row[40];
  const int bk = blockIdx.x, b = bk >> 2, k = bk & 3;
  const int d = threadIdx.x;                  // 192 threads

  float Areg[16], h[16], dtw[6];
#pragma unroll
  for (int n = 0; n < 16; ++n) {
    Areg[n] = -__expf(A_logs[((size_t)k * 192 + d) * 16 + n]);
    h[n] = 0.f;
  }
#pragma unroll
  for (int r = 0; r < 6; ++r) dtw[r] = dt_w[((size_t)k * 192 + d) * 6 + r];
  const float dtb = dt_b[k * 192 + d];
  const float Dsd = Ds[k * 192 + d];
  const float* xrow = &xc[(size_t)b * 192 * 2304 + (size_t)d * 2304];
  float* yrow = &ys[((size_t)bk * 192 + d) * 2304];

  for (int l = 0; l < 2304; ++l) {
    __syncthreads();
    if (d < 40) row[d] = x_dbl[((size_t)bk * 2304 + l) * 40 + d];
    __syncthreads();
    const float xv = xrow[scan_pos(k, l)];
    float draw = dtb;
#pragma unroll
    for (int r = 0; r < 6; ++r) draw += row[r] * dtw[r];
    const float delta = softplus_f(draw);
    const float dx = delta * xv;
    float y = 0.f;
#pragma unroll
    for (int n = 0; n < 16; ++n) {
      const float dA = __expf(delta * Areg[n]);
      h[n] = dA * h[n] + dx * row[6 + n];
      y += h[n] * row[22 + n];
    }
    yrow[l] = y + xv * Dsd;
  }
}

// ---------------- Kernel 5: merge 4 dirs + LN(192) + SiLU gate + out_proj ---
// xm[m][c] = x_res[m][c] + sum_d ynorm_gated[m][d] * Wout[c][d]
__global__ void k_merge_outproj(const float* __restrict__ ys,   // (8,192,2304)
                                const float* __restrict__ xz,   // z = ch 192..383
                                const float* __restrict__ on_g,
                                const float* __restrict__ on_b,
                                const float* __restrict__ wout, // (96,192)
                                const float* __restrict__ x,    // residual (B,96,2304)
                                float* __restrict__ xm) {       // (4608,96) NHWC
  __shared__ float Yr[16][192];
  __shared__ alignas(32) _Float16 Yh[16][192];
  __shared__ alignas(32) _Float16 Bs[96][192];  // N-major: Bs[c][d]
  __shared__ float s_mean[16], s_rinv[16];

  const int tid = threadIdx.x;                // 192 threads, 6 waves
  const int m0  = blockIdx.x * 16;            // 288

  for (int i = tid; i < 16 * 192; i += 192) {
    const int r = i / 192, dd = i % 192;
    const int m = m0 + r, b = m / 2304, l = m % 2304;
    const int hh = l / 48, ww = l % 48;
    const int p1 = ww * 48 + hh;
    const size_t base = (size_t)b * 4 * 192 * 2304 + (size_t)dd * 2304;
    const size_t dir = (size_t)192 * 2304;
    Yr[r][dd] = ys[base + l] + ys[base + 2 * dir + (2303 - l)] +
                ys[base + 1 * dir + p1] + ys[base + 3 * dir + (2303 - p1)];
  }
  __syncthreads();
  if (tid < 16) {
    float s = 0.f;
#pragma unroll 8
    for (int c = 0; c < 192; ++c) s += Yr[tid][c];
    const float mean = s * (1.0f / 192.0f);
    float v = 0.f;
#pragma unroll 8
    for (int c = 0; c < 192; ++c) { float d = Yr[tid][c] - mean; v += d * d; }
    s_mean[tid] = mean;
    s_rinv[tid] = rsqrtf(v * (1.0f / 192.0f) + EPSV);
  }
  __syncthreads();
  for (int i = tid; i < 16 * 192; i += 192) {
    const int r = i / 192, dd = i % 192;
    const int m = m0 + r;
    const float z = xz[(size_t)m * 384 + 192 + dd];
    const float yn = (Yr[r][dd] - s_mean[r]) * s_rinv[r] * on_g[dd] + on_b[dd];
    Yh[r][dd] = (_Float16)(yn * silu_f(z));
  }
  for (int i = tid; i < 96 * 192; i += 192) {  // straight copy: Wout is [c][d]
    const int c = i / 192, dd = i % 192;
    Bs[c][dd] = (_Float16)wout[(size_t)c * 192 + dd];
  }
  __syncthreads();

  const int w = tid >> 5;
  v8f acc = {};
#pragma unroll
  for (int kt = 0; kt < 6; ++kt) {
    v16h a  = load_a_frag(&Yh[0][kt * 32], 192);
    v16h bf = load_b_frag_t(&Bs[w * 16][kt * 32], 192);
    acc = wmma_f16(a, bf, acc);
  }
  const int lane = tid & 31, n = lane & 15, half = lane >> 4;
  const int c = w * 16 + n;
#pragma unroll
  for (int r = 0; r < 8; ++r) {
    const int m = m0 + half * 8 + r, b = m / 2304, l = m % 2304;
    xm[(size_t)m * 96 + c] = acc[r] + x[((size_t)b * 96 + c) * 2304 + l];
  }
}

// ---------------- Kernel 6: 1x1 stride-2 conv + BN + PReLU (channels 0..191)
__global__ void k_conv1x1_s2(const float* __restrict__ xm,   // (B,48,48,96) NHWC
                             const float* __restrict__ w1,   // (192,96)
                             const float* __restrict__ b1,
                             const float* __restrict__ bn_g,
                             const float* __restrict__ bn_b,
                             const float* __restrict__ bn_m,
                             const float* __restrict__ bn_v,
                             const float* __restrict__ prelu_a,
                             float* __restrict__ out) {      // (B,576,24,24)
  __shared__ alignas(32) _Float16 Ah[16][96];
  __shared__ alignas(32) _Float16 Bs[64][96];  // N-major
  const int tid = threadIdx.x;                // 128 threads
  const int m0  = blockIdx.x * 16;            // 72
  const int n0  = blockIdx.y * 64;            // 3

  for (int i = tid; i < 16 * 96; i += 128) {
    const int r = i / 96, c = i % 96;
    const int m = m0 + r, b = m / 576, rem = m % 576;
    const int oy = rem / 24, ox = rem % 24;
    Ah[r][c] = (_Float16)xm[((size_t)b * 2304 + (2 * oy) * 48 + 2 * ox) * 96 + c];
  }
  for (int i = tid; i < 64 * 96; i += 128) {   // straight copy: w1 is [o][c]
    const int j = i / 96, c = i % 96;
    Bs[j][c] = (_Float16)w1[(size_t)(n0 + j) * 96 + c];
  }
  __syncthreads();

  const int w = tid >> 5;
  v8f acc = {};
#pragma unroll
  for (int kt = 0; kt < 3; ++kt) {
    v16h a  = load_a_frag(&Ah[0][kt * 32], 96);
    v16h bf = load_b_frag_t(&Bs[w * 16][kt * 32], 96);
    acc = wmma_f16(a, bf, acc);
  }
  const int lane = tid & 31, n = lane & 15, half = lane >> 4;
  const int o = n0 + w * 16 + n;
  const float pa = prelu_a[0];
  const float sc = rsqrtf(bn_v[o] + EPSV);
#pragma unroll
  for (int r = 0; r < 8; ++r) {
    const int m = m0 + half * 8 + r, b = m / 576, sp = m % 576;
    float v = acc[r] + b1[o];
    v = (v - bn_m[o]) * sc * bn_g[o] + bn_b[o];
    out[((size_t)b * 576 + o) * 576 + sp] = (v > 0.f) ? v : pa * v;
  }
}

// ---------------- Kernel 7: generic 3x3 conv (im2col-by-tap WMMA) -----------
// mode 0: write NHWC f32 (bias only).  mode 1: bias+BN+PReLU -> NCHW out @chan0
__global__ void k_conv3x3(const float* __restrict__ src,   // NHWC (B,Hin,Hin,Cin)
                          int Hin, int Cin, int stride,
                          const float* __restrict__ wgt,   // (192,Cin,3,3)
                          const float* __restrict__ bias,
                          int mode, int chan0,
                          const float* __restrict__ bn_g,
                          const float* __restrict__ bn_b,
                          const float* __restrict__ bn_m,
                          const float* __restrict__ bn_v,
                          const float* __restrict__ prelu_a,
                          float* __restrict__ out_nchw,    // (B,576,24,24)
                          float* __restrict__ out_nhwc) {  // (B,24,24,192)
  __shared__ alignas(32) _Float16 Ah[16][32];
  __shared__ alignas(32) _Float16 Bs[64][32];  // N-major: Bs[o][c]
  const int tid = threadIdx.x;                // 128 threads
  const int m0  = blockIdx.x * 16;            // 72
  const int n0  = blockIdx.y * 64;            // 3
  const int kchunks = Cin / 32;

  const int w = tid >> 5;
  v8f acc = {};
  for (int tap = 0; tap < 9; ++tap) {
    const int ty = tap / 3, tx = tap % 3;
    for (int kc = 0; kc < kchunks; ++kc) {
      __syncthreads();
      for (int i = tid; i < 16 * 32; i += 128) {
        const int r = i / 32, cc = i % 32;
        const int m = m0 + r, b = m / 576, rem = m % 576;
        const int oy = rem / 24, ox = rem % 24;
        const int iy = oy * stride - 1 + ty, ix = ox * stride - 1 + tx;
        float v = 0.f;
        if (iy >= 0 && iy < Hin && ix >= 0 && ix < Hin)
          v = src[(((size_t)b * Hin + iy) * Hin + ix) * Cin + kc * 32 + cc];
        Ah[r][cc] = (_Float16)v;
      }
      for (int i = tid; i < 64 * 32; i += 128) {
        const int j = i / 32, cc = i % 32;
        Bs[j][cc] =
            (_Float16)wgt[((size_t)(n0 + j) * Cin + kc * 32 + cc) * 9 + tap];
      }
      __syncthreads();
      v16h a  = load_a_frag(&Ah[0][0], 32);
      v16h bf = load_b_frag_t(&Bs[w * 16][0], 32);
      acc = wmma_f16(a, bf, acc);
    }
  }
  const int lane = tid & 31, n = lane & 15, half = lane >> 4;
  const int o = n0 + w * 16 + n;
  if (mode == 1) {
    const int oc = chan0 + o;
    const float pa = prelu_a[0];
    const float sc = rsqrtf(bn_v[oc] + EPSV);
#pragma unroll
    for (int r = 0; r < 8; ++r) {
      const int m = m0 + half * 8 + r, b = m / 576, sp = m % 576;
      float v = acc[r] + bias[o];
      v = (v - bn_m[oc]) * sc * bn_g[oc] + bn_b[oc];
      out_nchw[((size_t)b * 576 + oc) * 576 + sp] = (v > 0.f) ? v : pa * v;
    }
  } else {
#pragma unroll
    for (int r = 0; r < 8; ++r) {
      const int m = m0 + half * 8 + r;
      out_nhwc[(size_t)m * 192 + o] = acc[r] + bias[o];
    }
  }
}

// ---------------- host orchestration ----------------------------------------
extern "C" void kernel_launch(void* const* d_in, const int* in_sizes, int n_in,
                              void* d_out, int out_size, void* d_ws, size_t ws_size,
                              hipStream_t stream) {
  const float* x      = (const float*)d_in[0];
  const float* ln_g   = (const float*)d_in[1];
  const float* ln_b   = (const float*)d_in[2];
  const float* wip    = (const float*)d_in[3];
  const float* dw     = (const float*)d_in[4];
  const float* db     = (const float*)d_in[5];
  const float* wxp    = (const float*)d_in[6];
  const float* dt_w   = (const float*)d_in[7];
  const float* dt_b   = (const float*)d_in[8];
  const float* A_logs = (const float*)d_in[9];
  const float* Ds     = (const float*)d_in[10];
  const float* on_g   = (const float*)d_in[11];
  const float* on_b   = (const float*)d_in[12];
  const float* wout   = (const float*)d_in[13];
  const float* w1     = (const float*)d_in[14];
  const float* b1     = (const float*)d_in[15];
  const float* w31    = (const float*)d_in[16];
  const float* b31    = (const float*)d_in[17];
  const float* w32    = (const float*)d_in[18];
  const float* b32    = (const float*)d_in[19];
  const float* w33    = (const float*)d_in[20];
  const float* b33    = (const float*)d_in[21];
  const float* bn_g   = (const float*)d_in[22];
  const float* bn_b   = (const float*)d_in[23];
  const float* bn_m   = (const float*)d_in[24];
  const float* bn_v   = (const float*)d_in[25];
  const float* pa     = (const float*)d_in[26];
  float* out = (float*)d_out;

  // workspace carve (floats)
  float* ws    = (float*)d_ws;
  float* xz    = ws;                      // 4608*384   = 1,769,472
  float* xc    = xz    + 1769472;         // 2*192*2304 =   884,736
  float* x_dbl = xc    + 884736;          // 8*2304*40  =   737,280
  float* ys    = x_dbl + 737280;          // 8*192*2304 = 3,538,944
  float* xm    = ys    + 3538944;         // 4608*96    =   442,368
  float* xtmp  = xm    + 442368;          // 2*24*24*192=   221,184

  k_ln_inproj<<<dim3(288, 6), 128, 0, stream>>>(x, ln_g, ln_b, wip, xz);
  k_dwconv_silu<<<dim3(3456), 256, 0, stream>>>(xz, dw, db, xc);
  k_xdbl<<<dim3(144, 8), 96, 0, stream>>>(xc, wxp, x_dbl);
  k_scan<<<dim3(8), 192, 0, stream>>>(xc, x_dbl, dt_w, dt_b, A_logs, Ds, ys);
  k_merge_outproj<<<dim3(288), 192, 0, stream>>>(ys, xz, on_g, on_b, wout, x, xm);
  // downsample head: channels [0,192) 1x1/s2, [192,384) 3x3/s2, [384,576) 3x3 s2 -> 3x3 s1
  k_conv1x1_s2<<<dim3(72, 3), 128, 0, stream>>>(xm, w1, b1, bn_g, bn_b, bn_m,
                                                bn_v, pa, out);
  k_conv3x3<<<dim3(72, 3), 128, 0, stream>>>(xm, 48, 96, 2, w31, b31, 1, 192,
                                             bn_g, bn_b, bn_m, bn_v, pa, out,
                                             nullptr);
  k_conv3x3<<<dim3(72, 3), 128, 0, stream>>>(xm, 48, 96, 2, w32, b32, 0, 0,
                                             nullptr, nullptr, nullptr, nullptr,
                                             nullptr, nullptr, xtmp);
  k_conv3x3<<<dim3(72, 3), 128, 0, stream>>>(xtmp, 24, 192, 1, w33, b33, 1, 384,
                                             bn_g, bn_b, bn_m, bn_v, pa, out,
                                             nullptr);
}
